// CompressiveMemory_57329223467206
// MI455X (gfx1250) — matrix operand order repaired and checked by reading, not compiled, not run on hardware
//
#include <hip/hip_runtime.h>

// ---------------------------------------------------------------------------
// Problem constants (from reference): B=4, S=8192, D=1024, H=16, dk=dv=64,
// SEG=256, NSEG=32.
// ---------------------------------------------------------------------------
#define BSZ   4
#define SLEN  8192
#define DMOD  1024
#define NH    16
#define DK    64
#define SEGL  256
#define NSEG  32
#define BS    (BSZ*SLEN)          // 32768 rows

typedef __attribute__((ext_vector_type(16))) __bf16 v16bf;
typedef __attribute__((ext_vector_type(8)))  __bf16 v8bf;
typedef __attribute__((ext_vector_type(8)))  float  v8f;
typedef __attribute__((ext_vector_type(4)))  int    v4i;

// Pointer types matching the async-DMA builtin signature:
//   (int4 addrspace(1)*, int4 addrspace(3)*, imm offset, imm cpol)
typedef __attribute__((address_space(1))) v4i* gv4i_p;
typedef __attribute__((address_space(3))) v4i* lv4i_p;

// ------------------------------ helpers ------------------------------------
// Native bf16 converts: gfx1250 has hardware v_cvt_*_bf16 paths.
__device__ inline float bf2f(__bf16 x) { return (float)x; }
__device__ inline __bf16 f2bf(float x) { return (__bf16)x; }

__device__ inline float elu1f(float x) {            // elu(x)+1
    return x > 0.f ? x + 1.f : __expf(x);
}

__device__ inline v8f wmma_bf16(v16bf a, v16bf b, v8f c) {
    return __builtin_amdgcn_wmma_f32_16x16x32_bf16(
        false, a, false, b, (short)0, c, false, false);
}

// A-operand fragment: 16(M) x 32(K) bf16, row-major source, row stride ld.
// Lane l: m = l&15, g = l>>4 ; element e<8 -> K=8g+e ; e>=8 -> K=16+8g+(e-8).
__device__ inline v16bf frag_a(const __bf16* base, int ld) {
    int ln = threadIdx.x & 31;
    int m = ln & 15, g = ln >> 4;
    const __bf16* p = base + m * ld + g * 8;
    v8bf lo = *(const v8bf*)p;
    v8bf hi = *(const v8bf*)(p + 16);
    v16bf r;
#pragma unroll
    for (int i = 0; i < 8; ++i) { r[i] = lo[i]; r[i + 8] = hi[i]; }
    return r;
}

// B-operand fragment: 32(K) x 16(N) bf16, from TRANSPOSED-B buffer Bt[n][k]
// (row stride ld). Lane l: n = l&15, g = l>>4 ; element e<8 -> K=8g+e ; e>=8 -> K=16+8g+(e-8).
__device__ inline v16bf frag_b(const __bf16* baseT, int ld) {
    int ln = threadIdx.x & 31;
    int n = ln & 15, g = ln >> 4;
    const __bf16* p = baseT + n * ld + g * 8;
    v8bf lo = *(const v8bf*)p;
    v8bf hi = *(const v8bf*)(p + 16);
    v16bf r;
#pragma unroll
    for (int i = 0; i < 8; ++i) { r[i] = lo[i]; r[i + 8] = hi[i]; }
    return r;
}

__device__ inline v16bf elu1v(v16bf a) {
    v16bf r;
#pragma unroll
    for (int e = 0; e < 16; ++e) r[e] = f2bf(elu1f(bf2f(a[e])));
    return r;
}

__device__ inline float rowmax16(float v) {
#pragma unroll
    for (int m = 1; m < 16; m <<= 1) v = fmaxf(v, __shfl_xor(v, m, 32));
    return v;
}
__device__ inline float rowsum16(float v) {
#pragma unroll
    for (int m = 1; m < 16; m <<= 1) v += __shfl_xor(v, m, 32);
    return v;
}

// ------------------------- kernel 0: f32 -> bf16 ---------------------------
__global__ void k_cvt(const float* __restrict__ src, __bf16* __restrict__ dst,
                      int n) {
    int i = blockIdx.x * 256 + threadIdx.x;
    if (i < n) dst[i] = f2bf(src[i]);
}

// ---------------- kernel 1: fused QKV projection (bf16 WMMA) ---------------
// C = x(32768x1024) @ [Wq|Wk|Wv](1024x3072) ; output scattered to
// Q/K/V[b][h][s][d] bf16. Block tile 128x64, 8 waves (4x2), wave tile 32x32.
// A tile moved with CDNA5 async global->LDS DMA (ASYNCcnt).
__global__ __launch_bounds__(256) void k_gemm_qkv(
    const __bf16* __restrict__ xh, const __bf16* __restrict__ Wqh,
    const __bf16* __restrict__ Wkh, const __bf16* __restrict__ Wvh,
    __bf16* __restrict__ Qh, __bf16* __restrict__ Kh, __bf16* __restrict__ Vh) {
    __shared__ __bf16 As[128 * 32];   // A tile, row-major
    __shared__ __bf16 Bt[64 * 32];    // B tile, transposed (n,k)

    int t = threadIdx.x;
    int m0 = blockIdx.x * 128;
    int nblk = blockIdx.y;            // 0..47
    int which = nblk >> 4;
    int n0 = (nblk & 15) * 64;
    const __bf16* W = which == 0 ? Wqh : (which == 1 ? Wkh : Wvh);
    __bf16* Out = which == 0 ? Qh : (which == 1 ? Kh : Vh);

    int wv = t >> 5, ln = t & 31;
    int wm = wv >> 1, wn = wv & 1;
    int g = ln >> 4, nl = ln & 15;

    v8f acc[2][2] = {};
    for (int k0 = 0; k0 < DMOD; k0 += 32) {
#pragma unroll
        for (int i = 0; i < 2; ++i) {               // 128x32 A tile, async DMA
            int e8 = (i * 256 + t) * 8;
            int r = e8 >> 5, c = e8 & 31;
            __builtin_amdgcn_global_load_async_to_lds_b128(
                (gv4i_p)&xh[(size_t)(m0 + r) * DMOD + k0 + c],
                (lv4i_p)&As[r * 32 + c], 0, 0);
        }
#pragma unroll
        for (int i = 0; i < 8; ++i) {               // 32x64 B tile, transposed
            int e = i * 256 + t;
            int kk = e >> 6, n = e & 63;
            Bt[n * 32 + kk] = W[(size_t)(k0 + kk) * DMOD + n0 + n];
        }
        asm volatile("s_wait_asynccnt 0" ::: "memory");
        __syncthreads();
        v16bf a0 = frag_a(&As[(wm * 32) * 32], 32);
        v16bf a1 = frag_a(&As[(wm * 32 + 16) * 32], 32);
        v16bf b0 = frag_b(&Bt[(wn * 32) * 32], 32);
        v16bf b1 = frag_b(&Bt[(wn * 32 + 16) * 32], 32);
        acc[0][0] = wmma_bf16(a0, b0, acc[0][0]);
        acc[0][1] = wmma_bf16(a0, b1, acc[0][1]);
        acc[1][0] = wmma_bf16(a1, b0, acc[1][0]);
        acc[1][1] = wmma_bf16(a1, b1, acc[1][1]);
        __syncthreads();
    }
#pragma unroll
    for (int sm = 0; sm < 2; ++sm)
#pragma unroll
        for (int sn = 0; sn < 2; ++sn)
#pragma unroll
            for (int r = 0; r < 8; ++r) {
                int M = m0 + wm * 32 + sm * 16 + r + 8 * g;
                int N = n0 + wn * 32 + sn * 16 + nl;
                int b = M >> 13, s = M & (SLEN - 1);
                int h = N >> 6, dd = N & 63;
                Out[(((size_t)(b * NH + h)) * SLEN + s) * DK + dd] =
                    f2bf(acc[sm][sn][r]);
            }
}

// ------------- kernel 2: per-(b,h) compressive-memory scan -----------------
__global__ __launch_bounds__(256) void k_attn(
    const __bf16* __restrict__ Qh, const __bf16* __restrict__ Kh,
    const __bf16* __restrict__ Vh, const float* __restrict__ betas,
    __bf16* __restrict__ atth) {
    extern __shared__ char smem[];
    __bf16* ks   = (__bf16*)smem;          // [256][64] key rows (scores Bt)
    __bf16* ksT  = ks  + SEGL * DK;        // [64][256] k transposed
    __bf16* vsT  = ksT + DK * SEGL;        // [64][256] v transposed
    float*  mem  = (float*)(vsT + DK * SEGL);  // [64][64] f32 state
    float*  z    = mem + DK * DK;              // [64]     f32 state
    __bf16* memT = (__bf16*)(z + DK);          // [64][64] bf16 mem^T mirror
    __bf16* pscr = memT + DK * DK;             // [8][16*32] per-wave P scratch

    int t = threadIdx.x, ln = t & 31, wv = t >> 5;
    int g = ln >> 4, nl = ln & 15;
    int bh = blockIdx.x;
    int b = bh >> 4, h = bh & 15;
    const size_t base = (size_t)bh * SLEN * DK;

    for (int i = t; i < DK * DK; i += 256) { mem[i] = 0.f; memT[i] = f2bf(0.f); }
    if (t < DK) z[t] = 1.0f / (float)DK;

    for (int seg = 0; seg < NSEG; ++seg) {
        __syncthreads();   // prev phase C done before overwriting tiles
        const __bf16* qg = Qh + base + (size_t)seg * SEGL * DK;
        const __bf16* kg = Kh + base + (size_t)seg * SEGL * DK;
        const __bf16* vg = Vh + base + (size_t)seg * SEGL * DK;
#pragma unroll
        for (int i = 0; i < 8; ++i) {
            int e8 = (i * 256 + t) * 8;
            v8bf kkv = *(const v8bf*)&kg[e8];
            *(v8bf*)&ks[e8] = kkv;
            v8bf vv = *(const v8bf*)&vg[e8];
            int l = e8 >> 6, d0 = e8 & 63;
#pragma unroll
            for (int e = 0; e < 8; ++e) {
                ksT[(d0 + e) * SEGL + l] = kkv[e];
                vsT[(d0 + e) * SEGL + l] = vv[e];
            }
        }
        if (seg + 1 < NSEG) {                   // pull next tiles toward L2
            __builtin_prefetch(qg + SEGL * DK, 0, 0);
            __builtin_prefetch(kg + SEGL * DK, 0, 0);
            __builtin_prefetch(vg + SEGL * DK, 0, 0);
        }
        __syncthreads();

        // ---- phase B: each wave handles two 16-row tiles ----
        for (int rti = 0; rti < 2; ++rti) {
            int m0 = (wv * 2 + rti) * 16;
            // q fragments straight from global (no cross-wave reuse)
            v16bf aq0 = frag_a(&qg[m0 * DK], DK);        // k = 0..31
            v16bf aq1 = frag_a(&qg[m0 * DK + 32], DK);   // k = 32..63
            v16bf sq0 = elu1v(aq0), sq1 = elu1v(aq1);    // sigma_q

            // numerator: sigma_q @ mem (16x64) via bf16 mem^T mirror
            v8f nacc[4] = {};
#pragma unroll
            for (int nt = 0; nt < 4; ++nt) {
                nacc[nt] = wmma_bf16(sq0, frag_b(&memT[(nt * 16) * DK], DK),
                                     nacc[nt]);
                nacc[nt] = wmma_bf16(sq1, frag_b(&memT[(nt * 16) * DK + 32], DK),
                                     nacc[nt]);
            }

            // denominator: sigma_q . z from the fragments themselves
            float part = 0.f;
#pragma unroll
            for (int e = 0; e < 16; ++e) {
                int K0 = 8 * g + e + (e >= 8 ? 8 : 0);   // frag_a K mapping
                part += bf2f(sq0[e]) * z[K0];
                part += bf2f(sq1[e]) * z[K0 + 32];
            }
            part += __shfl_xor(part, 16, 32);   // lane nl holds den[row=nl]

            // flash softmax over 256 keys in chunks of 32
            float mrow[8], srow[8];
#pragma unroll
            for (int r = 0; r < 8; ++r) { mrow[r] = -3.0e38f; srow[r] = 0.f; }
            v8f oacc[4] = {};
            __bf16* pw = pscr + wv * 16 * 32;

            for (int j = 0; j < 8; ++j) {
                v8f sc[2] = {};
#pragma unroll
                for (int c = 0; c < 2; ++c) {   // scores = q @ k^T, Bt == ks
                    const __bf16* kb = &ks[(j * 32 + c * 16) * DK];
                    sc[c] = wmma_bf16(aq0, frag_b(kb, DK),      sc[c]);
                    sc[c] = wmma_bf16(aq1, frag_b(kb + 32, DK), sc[c]);
                }
#pragma unroll
                for (int r = 0; r < 8; ++r) {
                    float s0 = sc[0][r] * 0.125f, s1 = sc[1][r] * 0.125f;
                    float nm = fmaxf(mrow[r], rowmax16(fmaxf(s0, s1)));
                    float alpha = __expf(mrow[r] - nm);
                    float e0 = __expf(s0 - nm), e1 = __expf(s1 - nm);
                    srow[r] = srow[r] * alpha + rowsum16(e0 + e1);
                    mrow[r] = nm;
                    int M = r + 8 * g;
                    pw[M * 32 + nl]      = f2bf(e0);
                    pw[M * 32 + 16 + nl] = f2bf(e1);
#pragma unroll
                    for (int nt = 0; nt < 4; ++nt) oacc[nt][r] *= alpha;
                }
                asm volatile("s_wait_dscnt 0" ::: "memory");   // P visible
                v16bf pa = frag_a(pw, 32);                     // P chunk 16x32
#pragma unroll
                for (int nt = 0; nt < 4; ++nt)                 // O += P @ V
                    oacc[nt] = wmma_bf16(
                        pa, frag_b(&vsT[(nt * 16) * SEGL + j * 32], SEGL),
                        oacc[nt]);
            }

            // combine + store with reference's raw (B,H,SEG,dv)->(B,SEG,H*dv)
#pragma unroll
            for (int nt = 0; nt < 4; ++nt) {
                int v = nt * 16 + nl;
                float bet = betas[h * DK + v];
                bet = 1.f / (1.f + __expf(-bet));
#pragma unroll
                for (int r = 0; r < 8; ++r) {
                    int M = r + 8 * g;
                    float den = __shfl(part, M, 32);
                    float av = bet * (nacc[nt][r] / den) +
                               (1.f - bet) * (oacc[nt][r] / srow[r]);
                    int lseq = m0 + M;
                    int r2 = h * 16 + (lseq >> 4);
                    int c2 = ((lseq & 15) << 6) + v;
                    atth[((size_t)(b * SLEN + seg * SEGL + r2)) * DMOD + c2] =
                        f2bf(av);
                }
            }
        }
        __syncthreads();   // all att_mem reads of mem/memT/z done

        // ---- phase C: mem += sigma_k^T @ v ; z += colsum(sigma_k) ----
#pragma unroll
        for (int ci = 0; ci < 2; ++ci) {
            int tile = wv * 2 + ci, ti = tile >> 2, tj = tile & 3;
            v8f acc;
#pragma unroll
            for (int r = 0; r < 8; ++r)
                acc[r] = mem[(ti * 16 + r + 8 * g) * DK + tj * 16 + nl];
            for (int j = 0; j < 8; ++j) {
                v16bf ak = elu1v(frag_a(&ksT[(ti * 16) * SEGL + j * 32], SEGL));
                v16bf bv = frag_b(&vsT[(tj * 16) * SEGL + j * 32], SEGL);
                acc = wmma_bf16(ak, bv, acc);
            }
#pragma unroll
            for (int r = 0; r < 8; ++r) {
                int kRow = ti * 16 + r + 8 * g;      // dk index
                int vCol = tj * 16 + nl;             // dv index
                mem[kRow * DK + vCol] = acc[r];
                memT[vCol * DK + kRow] = f2bf(acc[r]);   // bf16 mirror (B^T)
            }
        }
        {
            int d = t & 63, q4 = t >> 6;
            float p = 0.f;
            for (int l0 = 0; l0 < 64; ++l0)
                p += elu1f(bf2f(ksT[d * SEGL + q4 * 64 + l0]));
            atomicAdd(&z[d], p);     // ds_add_f32
        }
    }
}

// ------------------ kernel 3: out = att @ Wout (f32 out) -------------------
__global__ __launch_bounds__(256) void k_gemm_out(
    const __bf16* __restrict__ atth, const __bf16* __restrict__ Wouth,
    float* __restrict__ out) {
    __shared__ __bf16 As[128 * 32];
    __shared__ __bf16 Bt[64 * 32];
    int t = threadIdx.x;
    int m0 = blockIdx.x * 128;
    int n0 = blockIdx.y * 64;
    int wv = t >> 5, ln = t & 31;
    int wm = wv >> 1, wn = wv & 1;
    int g = ln >> 4, nl = ln & 15;

    v8f acc[2][2] = {};
    for (int k0 = 0; k0 < DMOD; k0 += 32) {
#pragma unroll
        for (int i = 0; i < 2; ++i) {               // A tile via async DMA
            int e8 = (i * 256 + t) * 8;
            int r = e8 >> 5, c = e8 & 31;
            __builtin_amdgcn_global_load_async_to_lds_b128(
                (gv4i_p)&atth[(size_t)(m0 + r) * DMOD + k0 + c],
                (lv4i_p)&As[r * 32 + c], 0, 0);
        }
#pragma unroll
        for (int i = 0; i < 8; ++i) {
            int e = i * 256 + t;
            int kk = e >> 6, n = e & 63;
            Bt[n * 32 + kk] = Wouth[(size_t)(k0 + kk) * DMOD + n0 + n];
        }
        asm volatile("s_wait_asynccnt 0" ::: "memory");
        __syncthreads();
        v16bf a0 = frag_a(&As[(wm * 32) * 32], 32);
        v16bf a1 = frag_a(&As[(wm * 32 + 16) * 32], 32);
        v16bf b0 = frag_b(&Bt[(wn * 32) * 32], 32);
        v16bf b1 = frag_b(&Bt[(wn * 32 + 16) * 32], 32);
        acc[0][0] = wmma_bf16(a0, b0, acc[0][0]);
        acc[0][1] = wmma_bf16(a0, b1, acc[0][1]);
        acc[1][0] = wmma_bf16(a1, b0, acc[1][0]);
        acc[1][1] = wmma_bf16(a1, b1, acc[1][1]);
        __syncthreads();
    }
#pragma unroll
    for (int sm = 0; sm < 2; ++sm)
#pragma unroll
        for (int sn = 0; sn < 2; ++sn)
#pragma unroll
            for (int r = 0; r < 8; ++r) {
                int M = m0 + wm * 32 + sm * 16 + r + 8 * g;
                int N = n0 + wn * 32 + sn * 16 + nl;
                out[(size_t)M * DMOD + N] = acc[sm][sn][r];
            }
}

// ------------------------------- launcher ----------------------------------
extern "C" void kernel_launch(void* const* d_in, const int* in_sizes, int n_in,
                              void* d_out, int out_size, void* d_ws,
                              size_t ws_size, hipStream_t stream) {
    const float* x     = (const float*)d_in[0];
    const float* Wq    = (const float*)d_in[1];
    const float* Wk    = (const float*)d_in[2];
    const float* Wv    = (const float*)d_in[3];
    const float* Wout  = (const float*)d_in[4];
    const float* betas = (const float*)d_in[5];
    float* out = (float*)d_out;

    const size_t NE_X = (size_t)BS * DMOD;       // 33,554,432
    const size_t NE_W = (size_t)DMOD * DMOD;     // 1,048,576
    const size_t NE_Q = (size_t)BSZ * NH * SLEN * DK;

    __bf16* wsb   = (__bf16*)d_ws;
    __bf16* xh    = wsb;               wsb += NE_X;
    __bf16* Wqh   = wsb;               wsb += NE_W;
    __bf16* Wkh   = wsb;               wsb += NE_W;
    __bf16* Wvh   = wsb;               wsb += NE_W;
    __bf16* Wouth = wsb;               wsb += NE_W;
    __bf16* Qh    = wsb;               wsb += NE_Q;
    __bf16* Kh    = wsb;               wsb += NE_Q;
    __bf16* Vh    = wsb;               wsb += NE_Q;
    __bf16* atth  = wsb;               wsb += NE_X;

    k_cvt<<<(int)((NE_X + 255) / 256), 256, 0, stream>>>(x, xh, (int)NE_X);
    k_cvt<<<(int)((NE_W + 255) / 256), 256, 0, stream>>>(Wq, Wqh, (int)NE_W);
    k_cvt<<<(int)((NE_W + 255) / 256), 256, 0, stream>>>(Wk, Wkh, (int)NE_W);
    k_cvt<<<(int)((NE_W + 255) / 256), 256, 0, stream>>>(Wv, Wvh, (int)NE_W);
    k_cvt<<<(int)((NE_W + 255) / 256), 256, 0, stream>>>(Wout, Wouth, (int)NE_W);

    dim3 g1(BS / 128, 48);
    k_gemm_qkv<<<g1, 256, 0, stream>>>(xh, Wqh, Wkh, Wvh, Qh, Kh, Vh);

    size_t lds = (size_t)(3 * SEGL * DK) * sizeof(__bf16)    // ks,ksT,vsT
               + (size_t)(DK * DK + DK) * sizeof(float)      // mem, z
               + (size_t)(DK * DK) * sizeof(__bf16)          // memT
               + (size_t)(8 * 16 * 32) * sizeof(__bf16);     // pscr
    k_attn<<<BSZ * NH, 256, lds, stream>>>(Qh, Kh, Vh, betas, atth);

    dim3 g2(BS / 128, 16);
    k_gemm_out<<<g2, 256, 0, stream>>>(atth, Wouth, out);
}
